// Paraboloid_47347719471158
// MI455X (gfx1250) — matrix operand order repaired and checked by reading, not compiled, not run on hardware
//
#include <hip/hip_runtime.h>
#include <stdint.h>

// Problem constants (match reference)
#define Bdim 16384
#define Ddim 1024
#define Mdim 4096

// GEMM tiling
#define BM 128
#define BN 128
#define BK 32
#define LDST 40   // LDS row stride in bf16 (32 + 8 pad -> 80B = 20 banks, conflict-free frags)
#define NSTAGE (Ddim / BK)

typedef __attribute__((ext_vector_type(16))) __bf16 v16bf;
typedef __attribute__((ext_vector_type(8)))  float  v8f;

__device__ __forceinline__ unsigned short f2bf_bits(float f) {
    union { float f; uint32_t u; } c; c.f = f;
    uint32_t u = c.u;
    uint32_t r = u + 0x7FFFu + ((u >> 16) & 1u);   // round-to-nearest-even
    return (unsigned short)(r >> 16);
}

// Async copy: 16 bytes global -> LDS, tracked by ASYNCcnt (CDNA5 path).
// GV mode: vdst = LDS byte address VGPR, vaddr = 64-bit global address pair.
__device__ __forceinline__ void async_ld_b128(uint32_t lds_off, const void* gaddr) {
    asm volatile("global_load_async_to_lds_b128 %0, %1, off"
                 :: "v"(lds_off), "v"(gaddr) : "memory");
}

// ---------------------------------------------------------------------------
// Pre-pass 1: x = 0.01*input -> bf16 workspace; x_sq[row] = sum(x*x) (fp32)
// ---------------------------------------------------------------------------
__global__ __launch_bounds__(256)
void prep_x(const float* __restrict__ in, unsigned short* __restrict__ xb,
            float* __restrict__ xsq) {
    __shared__ float red[256];
    const int row = blockIdx.x;
    const int t   = threadIdx.x;
    const float4 v = ((const float4*)(in + (size_t)row * Ddim))[t];
    const float x0 = 0.01f * v.x, x1 = 0.01f * v.y, x2 = 0.01f * v.z, x3 = 0.01f * v.w;
    ushort4 o;
    o.x = f2bf_bits(x0); o.y = f2bf_bits(x1); o.z = f2bf_bits(x2); o.w = f2bf_bits(x3);
    ((ushort4*)(xb + (size_t)row * Ddim))[t] = o;
    red[t] = x0*x0 + x1*x1 + x2*x2 + x3*x3;
    __syncthreads();
    for (int off = 128; off > 0; off >>= 1) {
        if (t < off) red[t] += red[t + off];
        __syncthreads();
    }
    if (t == 0) xsq[row] = red[0];
}

// ---------------------------------------------------------------------------
// Pre-pass 2: h,p -> bf16 workspace; p_sq[row] = sum(p*p) (fp32)
// ---------------------------------------------------------------------------
__global__ __launch_bounds__(256)
void prep_w(const float* __restrict__ pars, unsigned short* __restrict__ hb,
            unsigned short* __restrict__ pb, float* __restrict__ psq) {
    __shared__ float red[256];
    const int row = blockIdx.x;
    const int t   = threadIdx.x;
    const float* hrow = pars + (size_t)Mdim + (size_t)row * Ddim;
    const float* prow = pars + (size_t)Mdim + (size_t)Mdim * Ddim + (size_t)row * Ddim;

    const float4 hv = ((const float4*)hrow)[t];
    ushort4 ho;
    ho.x = f2bf_bits(hv.x); ho.y = f2bf_bits(hv.y);
    ho.z = f2bf_bits(hv.z); ho.w = f2bf_bits(hv.w);
    ((ushort4*)(hb + (size_t)row * Ddim))[t] = ho;

    const float4 pv = ((const float4*)prow)[t];
    ushort4 po;
    po.x = f2bf_bits(pv.x); po.y = f2bf_bits(pv.y);
    po.z = f2bf_bits(pv.z); po.w = f2bf_bits(pv.w);
    ((ushort4*)(pb + (size_t)row * Ddim))[t] = po;

    red[t] = pv.x*pv.x + pv.y*pv.y + pv.z*pv.z + pv.w*pv.w;
    __syncthreads();
    for (int off = 128; off > 0; off >>= 1) {
        if (t < off) red[t] += red[t + off];
        __syncthreads();
    }
    if (t == 0) psq[row] = red[0];
}

// ---------------------------------------------------------------------------
// Fused dual-GEMM + paraboloid epilogue, async double-buffered LDS pipeline.
// Block: 128x128 out tile; 256 threads = 8 waves (2x4). Wave: 64x32 subtile
// = 4x2 WMMA tiles per output matrix (h and p).
// ---------------------------------------------------------------------------
__global__ __launch_bounds__(256)
void paraboloid_gemm(const unsigned short* __restrict__ xb,
                     const unsigned short* __restrict__ hb,
                     const unsigned short* __restrict__ pb,
                     const float* __restrict__ xsq,
                     const float* __restrict__ psq,
                     const float* __restrict__ pars,
                     float* __restrict__ out) {
    __shared__ unsigned short sX[2][BM * LDST];
    __shared__ unsigned short sH[2][BN * LDST];
    __shared__ unsigned short sP[2][BN * LDST];
    __shared__ float xsq_s[BM];
    __shared__ float h0_s[BN];
    __shared__ float p0_s[BN];
    __shared__ float psq_s[BN];

    const int t    = threadIdx.x;
    const int wave = t >> 5;
    const int lane = t & 31;
    const int hi   = lane >> 4;     // half-wave select
    const int ln   = lane & 15;     // row/col within 16
    const int wm   = wave >> 2;     // 0..1  (wave row)
    const int wn   = wave & 3;      // 0..3  (wave col)
    const int rowBase = blockIdx.y * BM;
    const int colBase = blockIdx.x * BN;

    if (t < 128) {
        xsq_s[t] = xsq[rowBase + t];
        h0_s[t]  = pars[colBase + t];
        p0_s[t]  = pars[(size_t)Mdim + 2 * (size_t)Mdim * Ddim + colBase + t];
        psq_s[t] = psq[colBase + t];
    }

    v8f acc_h[4][2];
    v8f acc_p[4][2];
    const v8f zero = {0.f, 0.f, 0.f, 0.f, 0.f, 0.f, 0.f, 0.f};
#pragma unroll
    for (int mt = 0; mt < 4; ++mt)
#pragma unroll
        for (int nt = 0; nt < 2; ++nt) { acc_h[mt][nt] = zero; acc_p[mt][nt] = zero; }

    // Tile load map: 128 rows x 32 bf16 = 512 x 16B chunks per tile;
    // 2 chunks/thread/tile -> 6 async loads per thread per stage.
    const int lrow = t >> 2;             // 0..63 (and +64)
    const int kcB  = (t & 3) * 16;       // byte offset within 64B row payload
    const uint32_t sx0 = (uint32_t)(uintptr_t)(&sX[0][0]);
    const uint32_t sx1 = (uint32_t)(uintptr_t)(&sX[1][0]);
    const uint32_t sh0 = (uint32_t)(uintptr_t)(&sH[0][0]);
    const uint32_t sh1 = (uint32_t)(uintptr_t)(&sH[1][0]);
    const uint32_t sp0 = (uint32_t)(uintptr_t)(&sP[0][0]);
    const uint32_t sp1 = (uint32_t)(uintptr_t)(&sP[1][0]);

    const char* gx = (const char*)(xb + (size_t)rowBase * Ddim);
    const char* gh = (const char*)(hb + (size_t)colBase * Ddim);
    const char* gp = (const char*)(pb + (size_t)colBase * Ddim);

    // Issue all 6 async loads of one stage into buffer `buf`.
    auto issue_stage = [&](int stage, uint32_t bx, uint32_t bh_, uint32_t bp_) {
        const int k0B = stage * BK * 2;                 // byte offset in K
#pragma unroll
        for (int i = 0; i < 2; ++i) {
            const int row = lrow + i * 64;
            const uint32_t loff = (uint32_t)(row * (LDST * 2) + kcB);
            const size_t goff = (size_t)row * (Ddim * 2) + k0B + kcB;
            async_ld_b128(bx  + loff, gx + goff);
            async_ld_b128(bh_ + loff, gh + goff);
            async_ld_b128(bp_ + loff, gp + goff);
        }
    };

    union Frag { uint4 q[2]; v16bf v; };

    // Prologue: stage 0 into buffer 0.
    issue_stage(0, sx0, sh0, sp0);

    for (int it = 0; it < NSTAGE; ++it) {
        const int cur = it & 1;
        if (it + 1 < NSTAGE) {
            if (cur == 0) issue_stage(it + 1, sx1, sh1, sp1);
            else          issue_stage(it + 1, sx0, sh0, sp0);
            // allow the 6 just-issued loads to remain outstanding;
            // in-order completion => the previous stage's 6 are done
            asm volatile("s_wait_asynccnt 0x6" ::: "memory");
        } else {
            asm volatile("s_wait_asynccnt 0x0" ::: "memory");
        }
        __syncthreads();    // stage `it` resident in buf[cur] for all waves

        const unsigned short* sXc = sX[cur];
        const unsigned short* sHc = sH[cur];
        const unsigned short* sPc = sP[cur];

        Frag a[4], bhf[2], bpf[2];
        // A fragments: contiguous k order (k = 16*hi + s), same as B:
        // lanes 0-15 hold K[0..15]; lanes 16-31 hold K[16..31]
#pragma unroll
        for (int mt = 0; mt < 4; ++mt) {
            const unsigned short* base = &sXc[(wm * 64 + mt * 16 + ln) * LDST + hi * 16];
            a[mt].q[0] = *(const uint4*)(base);
            a[mt].q[1] = *(const uint4*)(base + 8);
        }
        // B fragments: contiguous k order, same as A: lanes 0-15 hold K[0..15],
        // lanes 16-31 hold K[16..31] -> 32 contiguous bytes per lane
#pragma unroll
        for (int nt = 0; nt < 2; ++nt) {
            const unsigned short* bh2 = &sHc[(wn * 32 + nt * 16 + ln) * LDST + hi * 16];
            bhf[nt].q[0] = *(const uint4*)(bh2);
            bhf[nt].q[1] = *(const uint4*)(bh2 + 8);
            const unsigned short* bp2 = &sPc[(wn * 32 + nt * 16 + ln) * LDST + hi * 16];
            bpf[nt].q[0] = *(const uint4*)(bp2);
            bpf[nt].q[1] = *(const uint4*)(bp2 + 8);
        }
#pragma unroll
        for (int mt = 0; mt < 4; ++mt) {
#pragma unroll
            for (int nt = 0; nt < 2; ++nt) {
                acc_h[mt][nt] = __builtin_amdgcn_wmma_f32_16x16x32_bf16(
                    false, a[mt].v, false, bhf[nt].v, (short)0, acc_h[mt][nt],
                    false, false);
                acc_p[mt][nt] = __builtin_amdgcn_wmma_f32_16x16x32_bf16(
                    false, a[mt].v, false, bpf[nt].v, (short)0, acc_p[mt][nt],
                    false, false);
            }
        }
        __syncthreads();    // all waves done reading buf[cur] before it is refilled
    }

    // Epilogue: out = 0.1 * ((hs+h0)^2 - (x_sq - 2*xp + p_sq) + p0)
    // C/D layout: lane 0-15 -> N=ln, VGPR r -> M=r ; lane 16-31 -> M=r+8
#pragma unroll
    for (int mt = 0; mt < 4; ++mt) {
#pragma unroll
        for (int nt = 0; nt < 2; ++nt) {
            const int colL = wn * 32 + nt * 16 + ln;
            const int col  = colBase + colL;
            const float h0c  = h0_s[colL];
            const float p0c  = p0_s[colL];
            const float psqc = psq_s[colL];
#pragma unroll
            for (int r = 0; r < 8; ++r) {
                const int rowL = wm * 64 + mt * 16 + r + hi * 8;
                const float hs = acc_h[mt][nt][r] + h0c;
                const float xp = acc_p[mt][nt][r];
                const float val =
                    0.1f * (hs * hs - (xsq_s[rowL] - 2.0f * xp + psqc) + p0c);
                out[(size_t)(rowBase + rowL) * Mdim + col] = val;
            }
        }
    }
}

// ---------------------------------------------------------------------------
extern "C" void kernel_launch(void* const* d_in, const int* in_sizes, int n_in,
                              void* d_out, int out_size, void* d_ws, size_t ws_size,
                              hipStream_t stream) {
    const float* input = (const float*)d_in[0];
    const float* pars  = (const float*)d_in[1];
    float* out = (float*)d_out;

    char* ws = (char*)d_ws;
    size_t off = 0;
    unsigned short* xb = (unsigned short*)(ws + off); off += (size_t)Bdim * Ddim * 2;
    unsigned short* hb = (unsigned short*)(ws + off); off += (size_t)Mdim * Ddim * 2;
    unsigned short* pb = (unsigned short*)(ws + off); off += (size_t)Mdim * Ddim * 2;
    float* xsq = (float*)(ws + off); off += (size_t)Bdim * 4;
    float* psq = (float*)(ws + off); off += (size_t)Mdim * 4;
    (void)ws_size; (void)in_sizes; (void)n_in; (void)out_size;

    prep_x<<<Bdim, 256, 0, stream>>>(input, xb, xsq);
    prep_w<<<Mdim, 256, 0, stream>>>(pars, hb, pb, psq);

    dim3 grid(Mdim / BN, Bdim / BM);
    paraboloid_gemm<<<grid, 256, 0, stream>>>(xb, hb, pb, xsq, psq, pars, out);
}
